// DyHetGNN_14001593385521
// MI455X (gfx1250) — matrix operand order repaired and checked by reading, not compiled, not run on hardware
//
#include <hip/hip_runtime.h>

#define D 256

typedef __bf16 bf16_t;
typedef bf16_t v16bf __attribute__((ext_vector_type(16)));
typedef float  v8f   __attribute__((ext_vector_type(8)));

union Frag { v16bf v; uint4 q[2]; };

__device__ __forceinline__ unsigned short f2bf(float f) {
    unsigned int u = __float_as_uint(f);
    u += 0x7FFFu + ((u >> 16) & 1u);          // round-to-nearest-even
    return (unsigned short)(u >> 16);
}

__device__ __forceinline__ unsigned int pack2bf(float lo, float hi) {
    return (unsigned int)f2bf(lo) | ((unsigned int)f2bf(hi) << 16);
}

__device__ __forceinline__ v8f vzero8() {
    v8f z;
#pragma unroll
    for (int i = 0; i < 8; ++i) z[i] = 0.0f;
    return z;
}

// A operand: 16x32 bf16 tile from LDS row-major [16][256] (buf points at M-tile base).
// ISA A layout: lanes 0-15 -> M=lane, K=0..7 & 16..23 ; lanes 16-31 -> M=lane-16, K=8..15 & 24..31
__device__ __forceinline__ v16bf ldsA(const unsigned short* buf, int lane, int kc) {
    int row = lane & 15;
    int k0  = kc * 32 + ((lane >> 4) << 3);
    const uint4* p = reinterpret_cast<const uint4*>(buf + row * D + k0);
    Frag f;
    f.q[0] = p[0];      // K = k0 .. k0+7
    f.q[1] = p[2];      // K = k0+16 .. k0+23
    return f.v;
}

// B operand: 32x16 bf16 fragment, pre-swizzled in workspace: lane's 16 bf16 contiguous (32B)
__device__ __forceinline__ v16bf gloB(const unsigned short* wsm, int lane, int nt, int kc) {
    const uint4* p = reinterpret_cast<const uint4*>(wsm + ((((nt * 8 + kc) * 32) + lane) << 4));
    Frag f;
    f.q[0] = p[0];
    f.q[1] = p[1];
    return f.v;
}

__device__ __forceinline__ v8f wmma_bf16(v16bf a, v16bf b, v8f c) {
    return __builtin_amdgcn_wmma_f32_16x16x32_bf16(false, a, false, b, (short)0, c, false, false);
}

__device__ __forceinline__ float sigm(float x) { return 1.0f / (1.0f + __expf(-x)); }

// --- weight converter: fp32 -> bf16 in WMMA B-fragment order -----------------
// frag layout: idx = (((nt*8 + kc)*32 + lane)*16 + j)
//   lanes 0-15 : n = nt*16 + lane,     k = kc*32 + (j&7) + (j>>3)*16
//   lanes 16-31: n = nt*16 + lane-16,  k = kc*32 + 8 + (j&7) + (j>>3)*16
__global__ void dyhet_convert_w(const float* __restrict__ src, unsigned short* __restrict__ dst,
                                int ntiles, int rowStride, int transpose) {
    int idx = blockIdx.x * blockDim.x + threadIdx.x;
    if (idx >= ntiles * 4096) return;
    int j    = idx & 15;
    int lane = (idx >> 4) & 31;
    int kc   = (idx >> 9) & 7;
    int nt   = idx >> 12;
    int k = kc * 32 + ((lane >> 4) << 3) + (j & 7) + ((j >> 3) << 4);
    int n = nt * 16 + (lane & 15);
    long s = transpose ? ((long)n * rowStride + k) : ((long)k * rowStride + n);
    dst[idx] = f2bf(src[s]);
}

// --- main fused kernel: 32 edges (two 16-row M-tiles) per workgroup ----------
// 4 waves; each wave owns 4 N-tiles and both M-tiles -> every B fragment loaded
// once feeds two WMMAs (halves L2 B-traffic per FLOP).
__global__ __launch_bounds__(128, 1) void dyhet_main(
    const int* __restrict__ src, const int* __restrict__ dst,
    const int* __restrict__ etype, const float* __restrict__ dt,
    const float* __restrict__ hid, const float* __restrict__ cell,
    const float* __restrict__ b1s, const float* __restrict__ b2s,
    const float* __restrict__ b1g, const float* __restrict__ b2g,
    const float* __restrict__ p_w, const float* __restrict__ p_b,
    const float* __restrict__ q_w, const float* __restrict__ q_b,
    const float* __restrict__ Ws_b, const float* __restrict__ Wg_b,
    const float* __restrict__ b_lstm, const float* __restrict__ w_out,
    const float* __restrict__ b_out, const float* __restrict__ decay_w,
    const unsigned short* __restrict__ wsb, float* __restrict__ out, int M)
{
    __shared__ __align__(16) unsigned short A_hs[32 * D];      // h_s bf16      16KB
    __shared__ __align__(16) unsigned short A_hg[32 * D];      // h_g bf16      16KB
    __shared__ __align__(16) unsigned short T1[4][32 * D];     // relu layer1   64KB
    __shared__ __align__(16) unsigned short MSG[2][32 * D];    // msg_s/msg_g   32KB
    __shared__ float accA[32], aGate[32], dtv[32], scoreAcc[32];
    __shared__ int   etyp[32], srcI[32], dstI[32];
    unsigned short* X = T1[0];   // X aliases T1[0] (T1 dead after stage-2 barrier)

    const int tid   = threadIdx.x;
    const int lane  = tid & 31;
    const int wave  = tid >> 5;
    const int edge0 = blockIdx.x * 32;
    const int DD    = 65536;                 // elems per swizzled 256x256 matrix

    if (tid < 32) {
        int e = edge0 + tid;
        srcI[tid] = src[e]; dstI[tid] = dst[e];
        etyp[tid] = etype[e]; dtv[tid] = dt[e];
        accA[tid] = 0.0f; scoreAcc[tid] = 0.0f;
    }
    __syncthreads();

    // --- gather h_s / h_g rows (fp32 -> bf16 packed stores) ---
    {
        int half = tid >> 6;                 // 0/1
        int c4   = (tid & 63) << 2;          // column (x4 floats)
#pragma unroll
        for (int it = 0; it < 16; ++it) {
            int r = it * 2 + half;
            const float4 hs = *reinterpret_cast<const float4*>(hid + (long)srcI[r] * D + c4);
            const float4 hg = *reinterpret_cast<const float4*>(hid + (long)dstI[r] * D + c4);
            int o = r * D + c4;
            uint2 ps, pg;
            ps.x = pack2bf(hs.x, hs.y); ps.y = pack2bf(hs.z, hs.w);
            pg.x = pack2bf(hg.x, hg.y); pg.y = pack2bf(hg.z, hg.w);
            *reinterpret_cast<uint2*>(&A_hs[o]) = ps;
            *reinterpret_cast<uint2*>(&A_hg[o]) = pg;
        }
    }
    __syncthreads();

    const int nlocal = lane & 15;
    const int row0   = (lane >> 4) << 3;     // C/D layout: VGPR r -> local M = r + row0

    // --- stage 1: relu(h @ W1[e] + b1[e]) for {s,g} x {e0,e1} ---
    for (int t = 0; t < 4; ++t) {
        int nt = wave * 4 + t, ncol = nt * 16 + nlocal;
        for (int mat = 0; mat < 4; ++mat) {
            const unsigned short* A = (mat < 2) ? A_hs : A_hg;
            const unsigned short* B = wsb + ((mat < 2) ? mat : (4 + (mat - 2))) * DD;
            const float* bias = (mat < 2) ? (b1s + mat * D) : (b1g + (mat - 2) * D);
            v8f acc0 = vzero8(), acc1 = vzero8();
#pragma unroll
            for (int kc = 0; kc < 8; ++kc) {
                v16bf b = gloB(B, lane, nt, kc);
                acc0 = wmma_bf16(ldsA(A, lane, kc), b, acc0);
                acc1 = wmma_bf16(ldsA(A + 16 * D, lane, kc), b, acc1);
            }
            float bv = bias[ncol];
#pragma unroll
            for (int r = 0; r < 8; ++r) {
                float v0 = acc0[r] + bv; v0 = v0 > 0.0f ? v0 : 0.0f;
                float v1 = acc1[r] + bv; v1 = v1 > 0.0f ? v1 : 0.0f;
                T1[mat][(r + row0) * D + ncol]        = f2bf(v0);
                T1[mat][(16 + r + row0) * D + ncol]   = f2bf(v1);
            }
        }
    }
    __syncthreads();

    // --- stage 2: layer2 both etypes, per-row select, p/q dot accumulate ---
    for (int t = 0; t < 4; ++t) {
        int nt = wave * 4 + t, ncol = nt * 16 + nlocal;
        for (int side = 0; side < 2; ++side) {
            const unsigned short* A0 = T1[side * 2 + 0];
            const unsigned short* A1 = T1[side * 2 + 1];
            const unsigned short* B0 = wsb + (side ? 6 : 2) * DD;
            const unsigned short* B1 = B0 + DD;
            const float* b2 = side ? b2g : b2s;
            const float* pq = side ? q_w : p_w;
            v8f e0m0 = vzero8(), e0m1 = vzero8(), e1m0 = vzero8(), e1m1 = vzero8();
#pragma unroll
            for (int kc = 0; kc < 8; ++kc) {
                v16bf b0 = gloB(B0, lane, nt, kc);
                e0m0 = wmma_bf16(ldsA(A0, lane, kc), b0, e0m0);
                e0m1 = wmma_bf16(ldsA(A0 + 16 * D, lane, kc), b0, e0m1);
                v16bf b1 = gloB(B1, lane, nt, kc);
                e1m0 = wmma_bf16(ldsA(A1, lane, kc), b1, e1m0);
                e1m1 = wmma_bf16(ldsA(A1 + 16 * D, lane, kc), b1, e1m1);
            }
            float bv0 = b2[ncol], bv1 = b2[D + ncol], pw = pq[ncol];
#pragma unroll
            for (int r = 0; r < 8; ++r) {
                int m0 = r + row0, m1 = 16 + r + row0;
                float v0 = (etyp[m0] == 0) ? (e0m0[r] + bv0) : (e1m0[r] + bv1);
                float v1 = (etyp[m1] == 0) ? (e0m1[r] + bv0) : (e1m1[r] + bv1);
                atomicAdd(&accA[m0], v0 * pw);
                atomicAdd(&accA[m1], v1 * pw);
                MSG[side][m0 * D + ncol] = f2bf(v0);
                MSG[side][m1 * D + ncol] = f2bf(v1);
            }
        }
    }
    __syncthreads();

    if (tid < 32) aGate[tid] = sigm(accA[tid] + p_b[0] + q_b[0]);
    __syncthreads();

    // --- stage 3: x = a*(msg_s@Ws^T + b) + (1-a)*(msg_g@Wg^T + b) ---
    for (int t = 0; t < 4; ++t) {
        int nt = wave * 4 + t, ncol = nt * 16 + nlocal;
        v8f s0 = vzero8(), s1 = vzero8(), g0 = vzero8(), g1 = vzero8();
#pragma unroll
        for (int kc = 0; kc < 8; ++kc) {
            v16bf bS = gloB(wsb + 8 * DD, lane, nt, kc);
            s0 = wmma_bf16(ldsA(MSG[0], lane, kc), bS, s0);
            s1 = wmma_bf16(ldsA(MSG[0] + 16 * D, lane, kc), bS, s1);
            v16bf bG = gloB(wsb + 9 * DD, lane, nt, kc);
            g0 = wmma_bf16(ldsA(MSG[1], lane, kc), bG, g0);
            g1 = wmma_bf16(ldsA(MSG[1] + 16 * D, lane, kc), bG, g1);
        }
        float bs = Ws_b[ncol], bg = Wg_b[ncol];
#pragma unroll
        for (int r = 0; r < 8; ++r) {
            int m0 = r + row0, m1 = 16 + r + row0;
            float a0 = aGate[m0], a1 = aGate[m1];
            float x0 = a0 * (s0[r] + bs) + (1.0f - a0) * (g0[r] + bg);
            float x1 = a1 * (s1[r] + bs) + (1.0f - a1) * (g1[r] + bg);
            X[m0 * D + ncol] = f2bf(x0);
            X[m1 * D + ncol] = f2bf(x1);
        }
    }
    __syncthreads();

    // --- stage 4: gates = X@Wx + h_g@Wh + b ; LSTM pointwise from registers ---
    const unsigned short* WxB = wsb + 10 * DD;   // 64 n-tiles
    const unsigned short* WhB = wsb + 14 * DD;   // 64 n-tiles
    float dw = fabsf(decay_w[0]);
    float* out_h = out + M;
    float* out_c = out + M + (long)M * D;
    for (int t = 0; t < 4; ++t) {
        int nt = wave * 4 + t;                   // state column tile [0,16)
        int ncol = nt * 16 + nlocal;
        v8f acc0[4], acc1[4];                    // i,f,g,o per M-tile
#pragma unroll
        for (int p = 0; p < 4; ++p) {
            int gnt = p * 16 + nt;               // gates n-tile in [0,64)
            v8f a0 = vzero8(), a1 = vzero8();
#pragma unroll
            for (int kc = 0; kc < 8; ++kc) {
                v16bf bX = gloB(WxB, lane, gnt, kc);
                a0 = wmma_bf16(ldsA(X, lane, kc), bX, a0);
                a1 = wmma_bf16(ldsA(X + 16 * D, lane, kc), bX, a1);
                v16bf bH = gloB(WhB, lane, gnt, kc);
                a0 = wmma_bf16(ldsA(A_hg, lane, kc), bH, a0);
                a1 = wmma_bf16(ldsA(A_hg + 16 * D, lane, kc), bH, a1);
            }
            float bl = b_lstm[p * D + ncol];
#pragma unroll
            for (int r = 0; r < 8; ++r) { a0[r] += bl; a1[r] += bl; }
            acc0[p] = a0; acc1[p] = a1;
        }
        float wo = w_out[ncol];
#pragma unroll
        for (int mt = 0; mt < 2; ++mt) {
#pragma unroll
            for (int r = 0; r < 8; ++r) {
                int m = mt * 16 + r + row0;
                float iv = mt ? acc1[0][r] : acc0[0][r];
                float fv = mt ? acc1[1][r] : acc0[1][r];
                float gv = mt ? acc1[2][r] : acc0[2][r];
                float ov = mt ? acc1[3][r] : acc0[3][r];
                float cg  = cell[(long)dstI[m] * D + ncol];
                float dec = cg * __expf(-dw * dtv[m]);
                float cn = sigm(fv) * dec + sigm(iv) * tanhf(gv);
                float hn = sigm(ov) * tanhf(cn);
                long eo = (long)(edge0 + m) * D + ncol;
                out_h[eo] = hn;
                out_c[eo] = cn;
                atomicAdd(&scoreAcc[m], hn * wo);
            }
        }
    }
    __syncthreads();
    if (tid < 32) out[edge0 + tid] = sigm(scoreAcc[tid] + b_out[0]);
}

extern "C" void kernel_launch(void* const* d_in, const int* in_sizes, int n_in,
                              void* d_out, int out_size, void* d_ws, size_t ws_size,
                              hipStream_t stream) {
    const int*   src    = (const int*)d_in[0];
    const int*   dstp   = (const int*)d_in[1];
    const int*   etype  = (const int*)d_in[2];
    const float* dt     = (const float*)d_in[3];
    const float* hid    = (const float*)d_in[4];
    const float* cell   = (const float*)d_in[5];
    const float* W1s    = (const float*)d_in[6];
    const float* b1s    = (const float*)d_in[7];
    const float* W2s    = (const float*)d_in[8];
    const float* b2s    = (const float*)d_in[9];
    const float* W1g    = (const float*)d_in[10];
    const float* b1g    = (const float*)d_in[11];
    const float* W2g    = (const float*)d_in[12];
    const float* b2g    = (const float*)d_in[13];
    const float* p_w    = (const float*)d_in[14];
    const float* p_b    = (const float*)d_in[15];
    const float* q_w    = (const float*)d_in[16];
    const float* q_b    = (const float*)d_in[17];
    const float* Ws_w   = (const float*)d_in[18];
    const float* Ws_b   = (const float*)d_in[19];
    const float* Wg_w   = (const float*)d_in[20];
    const float* Wg_b   = (const float*)d_in[21];
    const float* Wx     = (const float*)d_in[22];
    const float* Wh     = (const float*)d_in[23];
    const float* b_lstm = (const float*)d_in[24];
    const float* w_out  = (const float*)d_in[25];
    const float* b_out  = (const float*)d_in[26];
    const float* decay  = (const float*)d_in[27];

    const int M  = in_sizes[0];
    const int DD = 65536;
    unsigned short* wsb = (unsigned short*)d_ws;

    auto cw = [&](const float* s, unsigned short* d, int ntiles, int stride, int tr) {
        int total = ntiles * 4096;
        dyhet_convert_w<<<(total + 255) / 256, 256, 0, stream>>>(s, d, ntiles, stride, tr);
    };
    // rel-MLP weights: B[k][n] = W[e][k][n]
    cw(W1s,         wsb + 0 * DD, 16, 256, 0);
    cw(W1s + 65536, wsb + 1 * DD, 16, 256, 0);
    cw(W2s,         wsb + 2 * DD, 16, 256, 0);
    cw(W2s + 65536, wsb + 3 * DD, 16, 256, 0);
    cw(W1g,         wsb + 4 * DD, 16, 256, 0);
    cw(W1g + 65536, wsb + 5 * DD, 16, 256, 0);
    cw(W2g,         wsb + 6 * DD, 16, 256, 0);
    cw(W2g + 65536, wsb + 7 * DD, 16, 256, 0);
    // combine linears apply W^T: B[k][n] = W[n][k]
    cw(Ws_w, wsb + 8 * DD, 16, 256, 1);
    cw(Wg_w, wsb + 9 * DD, 16, 256, 1);
    // LSTM: B[k][n] = W[k][n], n in [0,1024)
    cw(Wx, wsb + 10 * DD, 64, 1024, 0);
    cw(Wh, wsb + 14 * DD, 64, 1024, 0);

    dyhet_main<<<M / 32, 128, 0, stream>>>(
        src, dstp, etype, dt, hid, cell,
        b1s, b2s, b1g, b2g, p_w, p_b, q_w, q_b,
        Ws_b, Wg_b, b_lstm, w_out, b_out, decay,
        wsb, (float*)d_out, M);
}
